// GSM_15942918603139
// MI455X (gfx1250) — matrix operand order repaired and checked by reading, not compiled, not run on hardware
//
#include <hip/hip_runtime.h>
#include <math.h>

// ---------------------------------------------------------------------------
// GSM (gate-shift module) for MI455X / gfx1250, wave32.
//
// Pass 0a: BN scale/shift precompute (256 channels).
// Pass 0b: build WMMA B-fragments (weights) in the exact 16x16x32 bf16
//          per-lane register layout: 576 frags (8 t x 8 ch-chunks x 9 offsets).
// Pass 1 : grouped 3x3x3 conv -> tanh gate via v_wmma_f32_16x16x32_bf16.
//          D[16 pixels,16 (t_out,g)] += A[16x32 im2col bf16] x B[32x16 weights].
//          x staged to LDS as bf16 (BN+ReLU applied once), channel-contiguous
//          so each A fragment = 2x ds_load_b128; B fragment = 2x global b128.
// Pass 2 : memory-bound elementwise gating + temporal shift + channel shuffle,
//          float4 coalesced (~410 MB HBM, the real cost at 23.3 TB/s).
// ---------------------------------------------------------------------------

typedef __attribute__((ext_vector_type(8)))  float  v8f;
typedef __attribute__((ext_vector_type(16))) __bf16 v16bf;

#define N_CLIP 8
#define TT     8
#define CH     256
#define HH     56
#define WW     56
#define PIX    (HH * WW)          // 3136
#define RB_ROWS 4                 // rows per workgroup in pass 1
#define NRB     (HH / RB_ROWS)    // 14 row blocks
#define STG_ROWS 6                // RB_ROWS + 2 halo
#define STG_COLS 58               // WW + 2 halo
#define CPAD   40                 // 32 channels padded to 40 halves (16B-align runs)
#define STG_N  (STG_ROWS * STG_COLS * CPAD)   // 13920 ushorts = 27.2 KB

// d_ws layout (bytes)
#define WS_GATE   0               // float[8*8*2*3136]  = 1,605,632 B
#define WS_SCALE  1605632         // float[256]
#define WS_SHIFT  1606656         // float[256]
#define WS_BFRAG  1607680         // ushort[576*512]    = 589,824 B

__device__ __forceinline__ unsigned short f2bf(float f) {
    unsigned int u = __float_as_uint(f);
    u += 0x7FFFu + ((u >> 16) & 1u);          // round-to-nearest-even
    return (unsigned short)(u >> 16);
}

// ---------------- Pass 0a: BN affine precompute ----------------
__global__ void bn_prep(const float* __restrict__ gamma, const float* __restrict__ beta,
                        const float* __restrict__ mean,  const float* __restrict__ var,
                        float* __restrict__ scale, float* __restrict__ shift) {
    int c = threadIdx.x;                       // 256 threads
    float s = gamma[c] * rsqrtf(var[c] + 1e-5f);
    scale[c] = s;
    shift[c] = beta[c] - mean[c] * s;
}

// ---------------- Pass 0b: B-fragment table ----------------
// Frag (t, cb, off): B[k, col]; k = channel-in-chunk (32), col = t_out*2 + g.
// bf16 B layout (32x16): lane L holds column N = L%16; half h holds
// K = (L<16 ? 0 : 8) + (h<8 ? h : 8 + h).  Stored as bfrag[frag*512 + lane*16 + h].
__global__ void build_bfrag(const float* __restrict__ w, unsigned short* __restrict__ bfrag) {
    int frag = blockIdx.x;                     // 576 blocks
    int off = frag % 9, cb = (frag / 9) & 7, t = frag / 72;
    int dh = off / 3, dw = off % 3;
    for (int e = threadIdx.x; e < 512; e += 256) {
        int lane = e >> 4, h = e & 15;
        int k    = ((lane & 16) ? 8 : 0) + (h < 8 ? h : 8 + h);
        int col  = lane & 15;
        int tout = col >> 1, g = col & 1;
        int c    = cb * 32 + k;
        int cgrp = c >> 7, cig = c & 127;
        int dt   = t - tout + 1;
        float val = 0.0f;
        if (cgrp == g && dt >= 0 && dt <= 2)
            val = w[(((g * 128 + cig) * 3 + dt) * 3 + dh) * 3 + dw];
        bfrag[frag * 512 + e] = f2bf(val);
    }
}

// ---------------- Pass 1: WMMA conv -> gate ----------------
// Grid 112 = 8 clips x 14 row blocks; block = 224 threads = 7 waves.
// Each wave owns 2 tiles of 16 linear pixels within the 4x56 row block.
__global__ __launch_bounds__(224) void conv_gate(
    const float* __restrict__ x, const float* __restrict__ bnscale,
    const float* __restrict__ bnshift, const unsigned short* __restrict__ bfrag,
    const float* __restrict__ conv_b, float* __restrict__ gate)
{
    __shared__ __attribute__((aligned(16))) unsigned short stage[STG_N];
    __shared__ float s_scale[CH];
    __shared__ float s_shift[CH];

    const int tid  = threadIdx.x;
    const int lane = tid & 31;
    const int wv   = tid >> 5;                 // 0..6
    const int nclip = blockIdx.x / NRB;
    const int rb    = blockIdx.x % NRB;

    for (int i = tid; i < CH; i += 224) { s_scale[i] = bnscale[i]; s_shift[i] = bnshift[i]; }
    for (int i = tid; i < STG_N; i += 224) stage[i] = 0;   // zero halos / OOB rows once

    // Per-lane invariants. A layout (16x32 bf16): lane holds row M = lane%16;
    // half-K runs start at (lane<16 ? 0 : 8) and (..)+16.
    const int aoff = (lane & 16) ? 8 : 0;
    const int m    = lane & 15;
    int hl[2], wl[2];
    {
        int p0 = (wv * 2) * 16 + m;            // tile 0 pixel (linear in 4x56 block)
        int p1 = p0 + 16;                      // tile 1 pixel
        hl[0] = p0 / WW; wl[0] = p0 % WW;
        hl[1] = p1 / WW; wl[1] = p1 % WW;
    }

    v8f acc0 = {};
    v8f acc1 = {};
    const float bias0 = conv_b[0], bias1 = conv_b[1];

    for (int t = 0; t < TT; ++t) {
        const int nt = nclip * TT + t;
        for (int cb = 0; cb < 8; ++cb) {
            __syncthreads();                   // compute of previous chunk done
            // ---- stage 32 channels x 6 rows x 56 cols, BN+ReLU, bf16 ----
            for (int j = 0; j < 12; ++j) {
                int job = tid + j * 224;       // 2688 jobs
                int wq  = job % 14;            // 14 float4 column-quads
                int r2  = job / 14;
                int k   = r2 & 31;
                int r   = r2 >> 5;             // stage row 0..5
                int h   = rb * RB_ROWS + r - 1;
                if (h >= 0 && h < HH) {
                    int c = cb * 32 + k;
                    const float4 xv = *(const float4*)(x + (nt * CH + c) * PIX + h * WW + wq * 4);
                    float sc = s_scale[c], sh = s_shift[c];
                    float v0 = fmaxf(fmaf(xv.x, sc, sh), 0.0f);
                    float v1 = fmaxf(fmaf(xv.y, sc, sh), 0.0f);
                    float v2 = fmaxf(fmaf(xv.z, sc, sh), 0.0f);
                    float v3 = fmaxf(fmaf(xv.w, sc, sh), 0.0f);
                    int sb = (r * STG_COLS + wq * 4 + 1) * CPAD + k;
                    stage[sb          ] = f2bf(v0);
                    stage[sb + CPAD   ] = f2bf(v1);
                    stage[sb + 2*CPAD ] = f2bf(v2);
                    stage[sb + 3*CPAD ] = f2bf(v3);
                }
            }
            __syncthreads();
            // ---- 9 offsets x 2 tiles of WMMA ----
            const unsigned short* fb = bfrag + ((t * 8 + cb) * 9) * 512;
            for (int off = 0; off < 9; ++off) {
                union { uint4 u[2]; v16bf v; } B;
                const uint4* bp = (const uint4*)(fb + off * 512 + lane * 16);
                B.u[0] = bp[0];
                B.u[1] = bp[1];
                int dh = off / 3, dw = off % 3;

                {   // tile 0
                    int site = (hl[0] + dh) * STG_COLS + (wl[0] + dw);
                    int b0 = site * CPAD + aoff;
                    union { uint4 u[2]; v16bf v; } A;
                    A.u[0] = *(const uint4*)(&stage[b0]);
                    A.u[1] = *(const uint4*)(&stage[b0 + 16]);
                    acc0 = __builtin_amdgcn_wmma_f32_16x16x32_bf16(
                        false, A.v, false, B.v, (short)0, acc0, false, false);
                }
                {   // tile 1
                    int site = (hl[1] + dh) * STG_COLS + (wl[1] + dw);
                    int b0 = site * CPAD + aoff;
                    union { uint4 u[2]; v16bf v; } A;
                    A.u[0] = *(const uint4*)(&stage[b0]);
                    A.u[1] = *(const uint4*)(&stage[b0 + 16]);
                    acc1 = __builtin_amdgcn_wmma_f32_16x16x32_bf16(
                        false, A.v, false, B.v, (short)0, acc1, false, false);
                }
            }
        }
    }

    // ---- store D with fused bias+tanh. D layout: lane holds col N = lane%16,
    //      VGPR v holds row M = v + (lane<16 ? 0 : 8). ----
    const int col  = m;
    const int tout = col >> 1, g = col & 1;
    const float bias = g ? bias1 : bias0;
    const int mb = aoff;                       // (lane&16)?8:0
    const int tb0 = (wv * 2) * 16;
    float* gp = gate + ((nclip * TT + tout) * 2 + g) * PIX + rb * (RB_ROWS * WW);
    #pragma unroll
    for (int v = 0; v < 8; ++v) {
        gp[tb0      + mb + v] = tanhf(acc0[v] + bias);
        gp[tb0 + 16 + mb + v] = tanhf(acc1[v] + bias);
    }
}

// ---------------- Pass 2: gating + temporal shift + channel shuffle ----------------
// Grid 16384 = 64 nt x 256 c_out; block 256 threads; float4 streaming.
__global__ __launch_bounds__(256) void gsm_apply(
    const float* __restrict__ x, const float* __restrict__ gate, float* __restrict__ out)
{
    const int blk   = blockIdx.x;
    const int nt    = blk >> 8;
    const int c_out = blk & 255;
    const int n = nt >> 3, t = nt & 7;

    int g, cx;
    if (c_out < 128) { g = 0; cx = ((c_out & 1) << 6) + (c_out >> 1); }
    else { int cp = c_out - 128; g = 1; cx = 128 + ((cp & 1) << 6) + (cp >> 1); }

    const float4* xc = (const float4*)(x    + (nt * CH + cx) * PIX);
    const float4* gc = (const float4*)(gate + ((n * TT + t) * 2 + g) * PIX);
    float4*       op = (float4*)(out + (nt * CH + c_out) * PIX);

    const float4* xo = nullptr;
    const float4* go = nullptr;
    bool has = false;
    if (g == 0) {
        if (t < 7) { has = true;
            xo = (const float4*)(x    + ((nt + 1) * CH + cx) * PIX);
            go = (const float4*)(gate + ((n * TT + t + 1) * 2 + g) * PIX); }
    } else {
        if (t > 0) { has = true;
            xo = (const float4*)(x    + ((nt - 1) * CH + cx) * PIX);
            go = (const float4*)(gate + ((n * TT + t - 1) * 2 + g) * PIX); }
    }

    for (int p = threadIdx.x; p < PIX / 4; p += 256) {
        float4 xv = xc[p], gv = gc[p];
        float4 r;
        r.x = xv.x - gv.x * xv.x;
        r.y = xv.y - gv.y * xv.y;
        r.z = xv.z - gv.z * xv.z;
        r.w = xv.w - gv.w * xv.w;
        if (has) {
            float4 x2 = xo[p], g2 = go[p];
            r.x = fmaf(g2.x, x2.x, r.x);
            r.y = fmaf(g2.y, x2.y, r.y);
            r.z = fmaf(g2.z, x2.z, r.z);
            r.w = fmaf(g2.w, x2.w, r.w);
        }
        op[p] = r;
    }
}

extern "C" void kernel_launch(void* const* d_in, const int* in_sizes, int n_in,
                              void* d_out, int out_size, void* d_ws, size_t ws_size,
                              hipStream_t stream) {
    (void)in_sizes; (void)n_in; (void)out_size; (void)ws_size;
    const float* x      = (const float*)d_in[0];
    const float* gamma  = (const float*)d_in[1];
    const float* beta   = (const float*)d_in[2];
    const float* mean   = (const float*)d_in[3];
    const float* var    = (const float*)d_in[4];
    const float* conv_w = (const float*)d_in[5];
    const float* conv_b = (const float*)d_in[6];
    float* out = (float*)d_out;

    char* ws = (char*)d_ws;
    float*          gate  = (float*)(ws + WS_GATE);
    float*          scale = (float*)(ws + WS_SCALE);
    float*          shift = (float*)(ws + WS_SHIFT);
    unsigned short* bfrag = (unsigned short*)(ws + WS_BFRAG);

    bn_prep    <<<1,   256, 0, stream>>>(gamma, beta, mean, var, scale, shift);
    build_bfrag<<<576, 256, 0, stream>>>(conv_w, bfrag);
    conv_gate  <<<N_CLIP * NRB, 224, 0, stream>>>(x, scale, shift, bfrag, conv_b, gate);
    gsm_apply  <<<64 * 256, 256, 0, stream>>>(x, gate, out);
}
